// GDN_Autoencoder_42116449305307
// MI455X (gfx1250) — matrix operand order repaired and checked by reading, not compiled, not run on hardware
//
#include <hip/hip_runtime.h>
#include <hip/hip_bf16.h>

typedef __attribute__((ext_vector_type(16))) _Float16 v16h;
typedef __attribute__((ext_vector_type(8)))  _Float16 v8h;
typedef __attribute__((ext_vector_type(8)))  float    v8f;

#define LRELU_SLOPE 0.01f

// ---------------------------------------------------------------------------
// f32 -> f16 elementwise convert (used once for h; A-operand fast path)
// ---------------------------------------------------------------------------
__global__ void f32_to_f16(const float* __restrict__ in,
                           _Float16* __restrict__ out, int n) {
    int i = blockIdx.x * blockDim.x + threadIdx.x;
    if (i < n) out[i] = (_Float16)in[i];
}

// ---------------------------------------------------------------------------
// Pack W [NOUT,K] f32 row-major  ->  Wt [K,NOUT] f16 row-major (transpose).
// Each lane's B fragment (16 k-values at one n = lane%16) is then a stride-NOUT gather.
// ---------------------------------------------------------------------------
__global__ void pack_wt(const float* __restrict__ W,
                        _Float16* __restrict__ Wt, int NOUT, int K) {
    int i = blockIdx.x * blockDim.x + threadIdx.x;
    if (i >= NOUT * K) return;
    int k = i / NOUT;
    int nn = i - k * NOUT;
    Wt[i] = (_Float16)W[(size_t)nn * K + k];
}

// ---------------------------------------------------------------------------
// Dual-output GEMM via WMMA:  C0 = A @ W0^T,  C1 = A @ W1^T
//   A16 : [M, K]    f16 row-major (pre-converted)
//   Wtx : [K, NOUT] f16 row-major (pre-transposed W)
//   Cx  : [M, NOUT] f32 row-major
// One wave per (16-row tile, W-select). Fragment packing per ISA 7.12.2:
//   A (16x32 f16): lane -> M = lane%16; lane-group hs = lane/16 holds
//     K = {hs*8..hs*8+7} and {16+hs*8..23+hs*8}  => two b128 loads.
//   B (32x16 f16): lane -> N = lane%16, K in same order as A => 16-element strided gather.
//   C/D (16x16 f32): VGPR r -> M = r + 8*hs, N = lane%16.
// ---------------------------------------------------------------------------
template<int K, int NOUT>
__global__ void gemm_dual_wmma(const _Float16* __restrict__ A16,
                               const _Float16* __restrict__ Wt0,
                               const _Float16* __restrict__ Wt1,
                               float* __restrict__ C0,
                               float* __restrict__ C1,
                               int M) {
    constexpr int NT = NOUT / 16;
    constexpr int KT = K / 32;

    const int lane = threadIdx.x & 31;
    const int wid  = blockIdx.x * (blockDim.x >> 5) + (threadIdx.x >> 5);
    const int tile = wid >> 1;
    const int row0 = tile * 16;
    if (row0 >= M) return;                       // wave-uniform exit

    const _Float16* __restrict__ Wt = (wid & 1) ? Wt1 : Wt0;
    float*          __restrict__ C  = (wid & 1) ? C1 : C0;

    __builtin_prefetch(Wt, 0, 1);                // global_prefetch_b8; weights L2-hot

    v8f acc[NT];
    #pragma unroll
    for (int i = 0; i < NT; ++i)
        #pragma unroll
        for (int r = 0; r < 8; ++r) acc[i][r] = 0.0f;

    const int m  = lane & 15;
    const int hs = lane >> 4;

    for (int kk = 0; kk < KT; ++kk) {
        // ---- A fragment: two contiguous 16B chunks per lane ----
        const _Float16* arow = A16 + (size_t)(row0 + m) * K + kk * 32;
        const v8h a0 = *(const v8h*)(arow + hs * 8);
        const v8h a1 = *(const v8h*)(arow + 16 + hs * 8);
        v16h a;
        #pragma unroll
        for (int i = 0; i < 8; ++i) { a[i] = a0[i]; a[i + 8] = a1[i]; }

        // ---- B fragments: lane -> N = lane%16, K in same order as A (strided gather) ----
        const _Float16* brow = Wt + (size_t)(kk * 32) * NOUT + m;
        #pragma unroll
        for (int nt = 0; nt < NT; ++nt) {
            v16h b;
            #pragma unroll
            for (int i = 0; i < 8; ++i) {
                b[i]     = brow[(size_t)(hs * 8 + i) * NOUT + nt * 16];
                b[i + 8] = brow[(size_t)(16 + hs * 8 + i) * NOUT + nt * 16];
            }
            acc[nt] = __builtin_amdgcn_wmma_f32_16x16x32_f16(
                false, a, false, b, (short)0, acc[nt], false, false);
        }
    }

    // ---- store C tile ----
    const int n  = lane & 15;
    const int mb = hs * 8;
    #pragma unroll
    for (int nt = 0; nt < NT; ++nt)
        #pragma unroll
        for (int r = 0; r < 8; ++r)
            C[(size_t)(row0 + mb + r) * NOUT + nt * 16 + n] = acc[nt][r];
}

// ---------------------------------------------------------------------------
// Per-node scalar: s[i] = Hd[i,:] . att  (linear-collapsed edge score)
// ---------------------------------------------------------------------------
__global__ void node_dot(const float* __restrict__ Hd,
                         const float* __restrict__ att,
                         float* __restrict__ s, int n, int dim) {
    int i = blockIdx.x * blockDim.x + threadIdx.x;
    if (i >= n) return;
    const float* r = Hd + (size_t)i * dim;
    float acc = 0.0f;
    for (int j = 0; j < dim; ++j) acc += r[j] * att[j];
    s[i] = acc;
}

// ---------------------------------------------------------------------------
// Encoder edge aggregation + ELU + hop-attention score, one hop.
// 64 threads per node (j = feature), 4 nodes per 256-thread block.
//   out[n,j] = elu( z[n,j] + sum_t a_t * hd[src_t, j] - hd[n,j] )
//   a_t = softmax_t( lrelu( s[src_t] - s[n] ) )
//   c[n, hop] = out[n,:] . att_comb
// Edges for node n are contiguous at [n*DEG, n*DEG+DEG) with dst == n.
// ---------------------------------------------------------------------------
template<int DEG>
__global__ void edge_agg_enc(const int* __restrict__ src,
                             const float* __restrict__ z,
                             const float* __restrict__ hd,
                             const float* __restrict__ s,
                             const float* __restrict__ att_comb,
                             float* __restrict__ Z,
                             float* __restrict__ c,
                             int hop, int n) {
    __shared__ float ls[4][DEG];
    __shared__ int   lsrc[4][DEG];
    __shared__ float red[4][64];

    const int j = threadIdx.x & 63;
    const int g = threadIdx.x >> 6;
    const int node = blockIdx.x * 4 + g;
    const bool active = node < n;

    if (active && j < DEG) {
        int sv = src[(size_t)node * DEG + j];
        lsrc[g][j] = sv;
        ls[g][j]   = s[sv];
    }
    __syncthreads();

    if (active) {
        const float sd = s[node];
        float e[DEG];
        float emax = -3.4e38f;
        #pragma unroll
        for (int t = 0; t < DEG; ++t) {
            float x = ls[g][t] - sd;
            x = (x > 0.0f) ? x : LRELU_SLOPE * x;
            e[t] = x;
            emax = fmaxf(emax, x);
        }
        float den = 0.0f;
        #pragma unroll
        for (int t = 0; t < DEG; ++t) { e[t] = __expf(e[t] - emax); den += e[t]; }
        const float inv = 1.0f / den;
        float acc = 0.0f;
        #pragma unroll
        for (int t = 0; t < DEG; ++t)
            acc += e[t] * hd[(size_t)lsrc[g][t] * 64 + j];   // coalesced row gather
        acc *= inv;

        float v = z[(size_t)node * 64 + j] + acc - hd[(size_t)node * 64 + j];
        v = (v > 0.0f) ? v : (__expf(v) - 1.0f);             // elu
        Z[(size_t)node * 64 + j] = v;
        red[g][j] = v * att_comb[j];
    }
    __syncthreads();
    if (active && j == 0) {
        float t = 0.0f;
        #pragma unroll
        for (int q = 0; q < 64; ++q) t += red[g][q];
        c[(size_t)node * 3 + hop] = t;
    }
}

// ---------------------------------------------------------------------------
// Hop-attention combine: a = softmax(lrelu(c[n,:])); zc = sum_k a_k * Z_k[n,:]
// Emits f16 directly (decoder GEMM A-operand fast path).
// ---------------------------------------------------------------------------
__global__ void hop_combine(const float* __restrict__ c,
                            const float* __restrict__ Z1,
                            const float* __restrict__ Z2,
                            const float* __restrict__ Z3,
                            _Float16* __restrict__ zc16, int n) {
    int idx = blockIdx.x * blockDim.x + threadIdx.x;
    if (idx >= n * 64) return;
    const int node = idx >> 6;
    float e0 = c[node * 3 + 0], e1 = c[node * 3 + 1], e2 = c[node * 3 + 2];
    e0 = (e0 > 0.0f) ? e0 : LRELU_SLOPE * e0;
    e1 = (e1 > 0.0f) ? e1 : LRELU_SLOPE * e1;
    e2 = (e2 > 0.0f) ? e2 : LRELU_SLOPE * e2;
    const float m = fmaxf(e0, fmaxf(e1, e2));
    const float w0 = __expf(e0 - m), w1 = __expf(e1 - m), w2 = __expf(e2 - m);
    const float inv = 1.0f / (w0 + w1 + w2);
    zc16[idx] = (_Float16)(inv * (w0 * Z1[idx] + w1 * Z2[idx] + w2 * Z3[idx]));
}

// ---------------------------------------------------------------------------
// Decoder edge aggregation (hop-3 graph, DEG=25, dim=128, no ELU) -> d_out
// 128 threads per node, 2 nodes per 256-thread block.
// ---------------------------------------------------------------------------
template<int DEG>
__global__ void edge_agg_dec(const int* __restrict__ src,
                             const float* __restrict__ z,
                             const float* __restrict__ hd,
                             const float* __restrict__ s,
                             float* __restrict__ out, int n) {
    __shared__ float ls[2][DEG];
    __shared__ int   lsrc[2][DEG];

    const int j = threadIdx.x & 127;
    const int g = threadIdx.x >> 7;
    const int node = blockIdx.x * 2 + g;
    const bool active = node < n;

    if (active && j < DEG) {
        int sv = src[(size_t)node * DEG + j];
        lsrc[g][j] = sv;
        ls[g][j]   = s[sv];
    }
    __syncthreads();
    if (!active) return;

    const float sd = s[node];
    float e[DEG];
    float emax = -3.4e38f;
    #pragma unroll
    for (int t = 0; t < DEG; ++t) {
        float x = ls[g][t] - sd;
        x = (x > 0.0f) ? x : LRELU_SLOPE * x;
        e[t] = x;
        emax = fmaxf(emax, x);
    }
    float den = 0.0f;
    #pragma unroll
    for (int t = 0; t < DEG; ++t) { e[t] = __expf(e[t] - emax); den += e[t]; }
    const float inv = 1.0f / den;
    float acc = 0.0f;
    #pragma unroll
    for (int t = 0; t < DEG; ++t)
        acc += e[t] * hd[(size_t)lsrc[g][t] * 128 + j];
    acc *= inv;

    out[(size_t)node * 128 + j] =
        z[(size_t)node * 128 + j] + acc - hd[(size_t)node * 128 + j];
}

// ---------------------------------------------------------------------------
extern "C" void kernel_launch(void* const* d_in, const int* in_sizes, int n_in,
                              void* d_out, int out_size, void* d_ws, size_t ws_size,
                              hipStream_t stream) {
    const float* h        = (const float*)d_in[0];   // [N,128]
    const float* fc_enc   = (const float*)d_in[1];   // [64,128]
    const float* diff_enc = (const float*)d_in[2];   // [64,128]
    const float* att_enc  = (const float*)d_in[3];   // [64]
    const float* fc_dec   = (const float*)d_in[4];   // [128,64]
    const float* diff_dec = (const float*)d_in[5];   // [128,64]
    const float* att_dec  = (const float*)d_in[6];   // [128]
    const float* att_comb = (const float*)d_in[7];   // [64]
    const int*   src1     = (const int*)d_in[8];     // dst arrays (9,11,13) unused:
    const int*   src2     = (const int*)d_in[10];    // edges are dst-grouped, dst==node
    const int*   src3     = (const int*)d_in[12];

    const int N = in_sizes[0] / 128;                 // 20000

    // ---- workspace carve: f32 region, then 16B-aligned f16 region ----
    float* ws = (float*)d_ws;
    size_t o = 0;
    float* z_enc  = ws + o; o += (size_t)N * 64;
    float* hd_enc = ws + o; o += (size_t)N * 64;
    float* s_buf  = ws + o; o += (size_t)N;
    float* Z1     = ws + o; o += (size_t)N * 64;
    float* Z2     = ws + o; o += (size_t)N * 64;
    float* Z3     = ws + o; o += (size_t)N * 64;
    float* cbuf   = ws + o; o += (size_t)N * 3;
    o = (o + 3) & ~(size_t)3;                        // 16-byte align the f16 region
    _Float16* hp = (_Float16*)(ws + o);
    size_t ho = 0;
    _Float16* h16      = hp + ho; ho += (size_t)N * 128;
    _Float16* zc16     = hp + ho; ho += (size_t)N * 64;
    _Float16* wt_enc0  = hp + ho; ho += 128 * 64;    // fc_enc^T   [K=128, NOUT=64]
    _Float16* wt_enc1  = hp + ho; ho += 128 * 64;    // diff_enc^T
    _Float16* wt_dec0  = hp + ho; ho += 64 * 128;    // fc_dec^T   [K=64, NOUT=128]
    _Float16* wt_dec1  = hp + ho; ho += 64 * 128;    // diff_dec^T
    // Reuse dead f32 regions for [N,128] decoder buffers (z_enc+hd_enc and
    // Z1+Z2 are contiguous N*128-float regions, dead by the time of writing).
    float* z_dec  = z_enc;
    float* hd_dec = Z1;
    float* s_dec  = s_buf;

    const int tiles = (N + 15) / 16;

    // 0) Operand packing: h -> f16, weights -> transposed f16
    f32_to_f16<<<((size_t)N * 128 + 255) / 256, 256, 0, stream>>>(h, h16, N * 128);
    pack_wt<<<(64 * 128 + 255) / 256, 256, 0, stream>>>(fc_enc,   wt_enc0, 64, 128);
    pack_wt<<<(64 * 128 + 255) / 256, 256, 0, stream>>>(diff_enc, wt_enc1, 64, 128);
    pack_wt<<<(128 * 64 + 255) / 256, 256, 0, stream>>>(fc_dec,   wt_dec0, 128, 64);
    pack_wt<<<(128 * 64 + 255) / 256, 256, 0, stream>>>(diff_dec, wt_dec1, 128, 64);

    // 1) Encoder GEMMs (shared across all 3 hops): z_enc, hd_enc
    {
        int waves  = 2 * tiles;
        int blocks = (waves + 7) / 8;
        gemm_dual_wmma<128, 64><<<blocks, 256, 0, stream>>>(
            h16, wt_enc0, wt_enc1, z_enc, hd_enc, N);
    }
    // 2) Per-node edge-score scalar s = hd_enc @ att_enc
    node_dot<<<(N + 255) / 256, 256, 0, stream>>>(hd_enc, att_enc, s_buf, N, 64);

    // 3-5) Edge softmax + aggregate + ELU + hop score, per hop
    edge_agg_enc<9 ><<<(N + 3) / 4, 256, 0, stream>>>(src1, z_enc, hd_enc, s_buf, att_comb, Z1, cbuf, 0, N);
    edge_agg_enc<17><<<(N + 3) / 4, 256, 0, stream>>>(src2, z_enc, hd_enc, s_buf, att_comb, Z2, cbuf, 1, N);
    edge_agg_enc<25><<<(N + 3) / 4, 256, 0, stream>>>(src3, z_enc, hd_enc, s_buf, att_comb, Z3, cbuf, 2, N);

    // 6) Hop-attention combine -> f16 directly
    hop_combine<<<((size_t)N * 64 + 255) / 256, 256, 0, stream>>>(cbuf, Z1, Z2, Z3, zc16, N);

    // 7) Decoder GEMMs: z_dec, hd_dec  (K=64, NOUT=128)
    {
        int waves  = 2 * tiles;
        int blocks = (waves + 7) / 8;
        gemm_dual_wmma<64, 128><<<blocks, 256, 0, stream>>>(
            zc16, wt_dec0, wt_dec1, z_dec, hd_dec, N);
    }
    // 8) s_dec = hd_dec @ att_dec
    node_dot<<<(N + 255) / 256, 256, 0, stream>>>(hd_dec, att_dec, s_dec, N, 128);

    // 9) Decoder edge aggregation on hop-3 graph -> d_out (no ELU)
    edge_agg_dec<25><<<(N + 1) / 2, 256, 0, stream>>>(
        src3, z_dec, hd_dec, s_dec, (float*)d_out, N);
}
